// ModularRoutingNetwork_85572928405667
// MI455X (gfx1250) — compile-verified, matrix operand order checked by reading, not run on hardware
//
#include <hip/hip_runtime.h>
#include <hip/hip_bf16.h>

// ---------------------------------------------------------------------------
// MoE (top-2 of 8 experts) for MI455X / gfx1250, wave32, bf16 WMMA.
//
// Pipeline (all deterministic, no float atomics):
//   1) k_cvt_x       : x f32 -> bf16
//   2) k_transpose   : w1 (E,D,H) -> w1T (E,H,D) bf16 ; w2 (E,H,O) -> w2T (E,O,H) bf16
//   3) k_route       : raw/softmax/top2/entropy, per-block (expert,k) histograms
//   4) k_scan        : serial prefix scan -> segment bases, per-block offsets,
//                      tile->expert map, pad-slot init
//   5) k_slot        : deterministic in-block rank -> slot assignment
//   6) k_moe pass0/1 : fused expert GEMM (x@w1 -> relu -> @w2), WMMA bf16,
//                      pass0 writes out, pass1 accumulates (each token appears
//                      exactly once per pass -> no collisions)
//   7) k_entf        : entropy finalize
// ---------------------------------------------------------------------------

#define B_ 16384
#define D_ 512
#define H_ 1024
#define O_ 512
#define E_ 8
#define K_ 2

#define SLOTCAP 16640              // B + 8*32 padding headroom per k
#define TILES_K 520                // SLOTCAP / 32
#define NBLK    64                 // B / 256 routing blocks

typedef unsigned short ushort_t;
typedef __attribute__((ext_vector_type(16))) __bf16 v16bf;
typedef __attribute__((ext_vector_type(8)))  float  v8f;

union FragBF { v16bf v; uint4 q[2]; };

__device__ __forceinline__ ushort_t f2bf(float f) {
    unsigned int u = __float_as_uint(f);
    unsigned int r = u + 0x7FFFu + ((u >> 16) & 1u);   // RNE
    return (ushort_t)(r >> 16);
}

// Load a 16x32 bf16 fragment (A as MxK row-major, or B as NxK row-major =
// transposed weight).  Per ISA 7.12.2 (16-bit A 16x32): lanes 0-15 -> row=lane,
// K = {0..7, 16..23}; lanes 16-31 -> row=lane-16, K = {8..15, 24..31}.
// base points at (tile_row0, k0); ld in elements (multiple of 8).
__device__ __forceinline__ v16bf ld_frag(const ushort_t* base, int ld) {
    int lane = threadIdx.x & 31;
    int row  = lane & 15;
    int kb   = (lane >> 4) * 8;
    const ushort_t* p = base + row * ld + kb;
    FragBF f;
    f.q[0] = *(const uint4*)(p);
    f.q[1] = *(const uint4*)(p + 16);
    return f.v;
}

__device__ __forceinline__ v8f wmma_bf16(v16bf a, v16bf b, v8f c) {
    return __builtin_amdgcn_wmma_f32_16x16x32_bf16(false, a, false, b,
                                                   (short)0, c, false, false);
}

// ---------------------------------------------------------------- convert x
__global__ __launch_bounds__(256) void k_cvt_x(const float* __restrict__ x,
                                               ushort_t* __restrict__ xb) {
    size_t i = ((size_t)blockIdx.x * 256 + threadIdx.x) * 4;
    if (i + 3 < (size_t)B_ * D_) {
        float4 v = *(const float4*)(x + i);
        xb[i+0] = f2bf(v.x); xb[i+1] = f2bf(v.y);
        xb[i+2] = f2bf(v.z); xb[i+3] = f2bf(v.w);
    }
}

// ------------------------------------------------- transpose (R,C) -> (C,R)
__global__ __launch_bounds__(256) void k_transpose(const float* __restrict__ in,
                                                   ushort_t* __restrict__ outp,
                                                   int R, int C) {
    __shared__ float tile[32][33];
    int tilesPerMat = (R / 32) * (C / 32);
    int b  = blockIdx.x;
    int e  = b / tilesPerMat;
    int t  = b % tilesPerMat;
    int tr = t / (C / 32);
    int tc = t % (C / 32);
    const float* src = in + (size_t)e * R * C;
    ushort_t*    dst = outp + (size_t)e * R * C;
    int rr = threadIdx.x >> 5;      // 0..7
    int cc = threadIdx.x & 31;      // 0..31
    #pragma unroll
    for (int j = 0; j < 4; j++) {
        int row = tr * 32 + rr + j * 8;
        tile[rr + j * 8][cc] = src[(size_t)row * C + tc * 32 + cc];
    }
    __syncthreads();
    #pragma unroll
    for (int j = 0; j < 4; j++) {
        int orow = tc * 32 + rr + j * 8;   // C dimension
        int ocol = tr * 32 + cc;           // R dimension
        dst[(size_t)orow * R + ocol] = f2bf(tile[cc][rr + j * 8]);
    }
}

// ---------------------------------------------------------------- routing
__global__ __launch_bounds__(256) void k_route(const float* __restrict__ x,
                                               const float* __restrict__ gw,
                                               const float* __restrict__ gb,
                                               float* __restrict__ rawOut,
                                               float* __restrict__ idxOut,
                                               int*   __restrict__ tokE,
                                               float* __restrict__ tokW,
                                               int*   __restrict__ blockCnt,
                                               float* __restrict__ blockEnt) {
    __shared__ int   cnt[16];
    __shared__ float se[256];
    int tid = threadIdx.x;
    int b   = blockIdx.x * 256 + tid;
    if (tid < 16) cnt[tid] = 0;
    __syncthreads();

    float acc[E_];
    #pragma unroll
    for (int e = 0; e < E_; e++) acc[e] = gb[e];
    const float* xr = x + (size_t)b * D_;
    for (int d = 0; d < D_; d++) {
        float xv = xr[d];
        const float* g = gw + d * E_;
        #pragma unroll
        for (int e = 0; e < E_; e++) acc[e] += xv * g[e];
    }
    #pragma unroll
    for (int e = 0; e < E_; e++) rawOut[(size_t)b * E_ + e] = acc[e];

    float m = acc[0];
    #pragma unroll
    for (int e = 1; e < E_; e++) m = fmaxf(m, acc[e]);
    float p[E_], s = 0.f;
    #pragma unroll
    for (int e = 0; e < E_; e++) { p[e] = __expf(acc[e] - m); s += p[e]; }
    float inv = 1.f / s;
    float ent = 0.f;
    #pragma unroll
    for (int e = 0; e < E_; e++) { p[e] *= inv; ent -= p[e] * logf(p[e] + 1e-9f); }

    int i0 = 0;
    #pragma unroll
    for (int e = 1; e < E_; e++) if (p[e] > p[i0]) i0 = e;
    int i1 = (i0 == 0) ? 1 : 0;
    #pragma unroll
    for (int e = 0; e < E_; e++) if (e != i0 && p[e] > p[i1]) i1 = e;

    float t  = __expf(p[i1] - p[i0]);    // softmax over (top1, top2)
    float c0 = 1.f / (1.f + t);
    float c1 = t * c0;

    idxOut[(size_t)b * 2 + 0] = (float)i0;
    idxOut[(size_t)b * 2 + 1] = (float)i1;
    tokE[b * 2 + 0] = i0;  tokE[b * 2 + 1] = i1;
    tokW[b * 2 + 0] = c0;  tokW[b * 2 + 1] = c1;

    atomicAdd(&cnt[0 * 8 + i0], 1);      // integer -> order-independent
    atomicAdd(&cnt[1 * 8 + i1], 1);

    se[tid] = ent;
    __syncthreads();
    for (int st = 128; st > 0; st >>= 1) {
        if (tid < st) se[tid] += se[tid + st];
        __syncthreads();
    }
    if (tid == 0) blockEnt[blockIdx.x] = se[0];
    if (tid < 16) blockCnt[blockIdx.x * 16 + tid] = cnt[tid];
}

// --------------------------------------------------------------- scan (tiny)
__global__ void k_scan(const int* __restrict__ blockCnt,
                       int* __restrict__ blockOff,
                       int* __restrict__ tileExp,
                       int* __restrict__ slotTok) {
    if (threadIdx.x != 0) return;
    for (int k = 0; k < 2; k++) {
        int base = k * SLOTCAP;
        int t = 0;
        for (int e = 0; e < E_; e++) {
            int tot = 0;
            for (int blk = 0; blk < NBLK; blk++)
                tot += blockCnt[blk * 16 + k * 8 + e];
            int run = base;
            for (int blk = 0; blk < NBLK; blk++) {
                blockOff[blk * 16 + k * 8 + e] = run;
                run += blockCnt[blk * 16 + k * 8 + e];
            }
            int pad = (tot + 31) & ~31;
            for (int s2 = tot; s2 < pad; s2++) slotTok[base + s2] = -1;
            int ntile = pad >> 5;
            for (int j = 0; j < ntile; j++) tileExp[k * TILES_K + t++] = e;
            base += pad;
        }
        while (t < TILES_K) tileExp[k * TILES_K + t++] = -1;
    }
}

// -------------------------------------------------------- slot assignment
__global__ __launch_bounds__(256) void k_slot(const int* __restrict__ tokE,
                                              const float* __restrict__ tokW,
                                              const int* __restrict__ blockOff,
                                              int*   __restrict__ slotTok,
                                              float* __restrict__ slotW) {
    __shared__ int le0[256], le1[256];
    int tid = threadIdx.x;
    int b   = blockIdx.x * 256 + tid;
    int e0  = tokE[b * 2 + 0];
    int e1  = tokE[b * 2 + 1];
    le0[tid] = e0; le1[tid] = e1;
    __syncthreads();
    int r0 = 0, r1 = 0;
    for (int j = 0; j < tid; j++) {      // deterministic in-block rank
        r0 += (le0[j] == e0);
        r1 += (le1[j] == e1);
    }
    int s0 = blockOff[blockIdx.x * 16 + 0 * 8 + e0] + r0;
    int s1 = blockOff[blockIdx.x * 16 + 1 * 8 + e1] + r1;
    slotTok[s0] = b;  slotW[s0] = tokW[b * 2 + 0];
    slotTok[s1] = b;  slotW[s1] = tokW[b * 2 + 1];
}

// ------------------------------------------------ fused expert GEMM (WMMA)
#define XS_LD 520   // 512 + 8 bf16 pad (16B aligned rows, bank-staggered)
#define HC_LD 264   // 256 + 8

__global__ __launch_bounds__(256) void k_moe(int pass,
                                             const int*      __restrict__ slotTok,
                                             const float*    __restrict__ slotW,
                                             const int*      __restrict__ tileExp,
                                             const ushort_t* __restrict__ xb,
                                             const ushort_t* __restrict__ w1T,
                                             const ushort_t* __restrict__ w2T,
                                             const float*    __restrict__ b1,
                                             const float*    __restrict__ b2,
                                             float*          __restrict__ out) {
    int tile = blockIdx.x;
    int e = tileExp[pass * TILES_K + tile];
    if (e < 0) return;
    int slotBase = pass * SLOTCAP + tile * 32;

    __shared__ ushort_t xs[32 * XS_LD];   // gathered x tile, bf16
    __shared__ ushort_t hc[32 * HC_LD];   // hidden chunk, bf16
    __shared__ int      stok[32];
    __shared__ float    swt[32];

    int tid  = threadIdx.x;
    int lane = tid & 31;
    int wave = tid >> 5;                  // 8 waves
    int half = lane >> 4;
    int nlo  = lane & 15;

    if (tid < 32) { stok[tid] = slotTok[slotBase + tid]; swt[tid] = slotW[slotBase + tid]; }

    // Cooperative gather of 32 x-rows (bf16) into LDS.
    {
        int r  = tid >> 3;                // 32 rows, 8 threads/row
        int l8 = tid & 7;
        int tok = slotTok[slotBase + r];
        if (tok < 0) tok = 0;
        const uint4* src = (const uint4*)(xb + (size_t)tok * D_);
        uint4* dst = (uint4*)(xs + r * XS_LD);
        #pragma unroll
        for (int j = 0; j < 8; j++) dst[l8 + j * 8] = src[l8 + j * 8];
    }
    __syncthreads();

    const ushort_t* w1e = w1T + (size_t)e * H_ * D_;   // (H, D) bf16
    const ushort_t* w2e = w2T + (size_t)e * O_ * H_;   // (O, H) bf16

    v8f yacc[2][4];
    #pragma unroll
    for (int mt = 0; mt < 2; mt++)
        #pragma unroll
        for (int nt = 0; nt < 4; nt++) { v8f z = {}; yacc[mt][nt] = z; }

    for (int hc0 = 0; hc0 < H_; hc0 += 256) {
        // ---- stage 1: h[32 x 256] chunk; wave w owns cols [w*32, w*32+32)
        v8f ha[2][2];
        #pragma unroll
        for (int mt = 0; mt < 2; mt++)
            #pragma unroll
            for (int nt = 0; nt < 2; nt++) { v8f z = {}; ha[mt][nt] = z; }

        for (int k0 = 0; k0 < D_; k0 += 32) {
            v16bf a0 = ld_frag(xs + 0  * XS_LD + k0, XS_LD);
            v16bf a1 = ld_frag(xs + 16 * XS_LD + k0, XS_LD);
            v16bf b0 = ld_frag(w1e + (size_t)(hc0 + wave * 32 + 0 ) * D_ + k0, D_);
            v16bf bb = ld_frag(w1e + (size_t)(hc0 + wave * 32 + 16) * D_ + k0, D_);
            ha[0][0] = wmma_bf16(a0, b0, ha[0][0]);
            ha[0][1] = wmma_bf16(a0, bb, ha[0][1]);
            ha[1][0] = wmma_bf16(a1, b0, ha[1][0]);
            ha[1][1] = wmma_bf16(a1, bb, ha[1][1]);
        }
        #pragma unroll
        for (int mt = 0; mt < 2; mt++)
            #pragma unroll
            for (int nt = 0; nt < 2; nt++) {
                int col = wave * 32 + nt * 16 + nlo;
                float bv = b1[(size_t)e * H_ + hc0 + col];
                #pragma unroll
                for (int i = 0; i < 8; i++) {
                    int m = mt * 16 + half * 8 + i;
                    float v = ha[mt][nt][i] + bv;
                    v = fmaxf(v, 0.f);
                    hc[m * HC_LD + col] = f2bf(v);
                }
            }
        __syncthreads();

        // ---- stage 2: yacc += h_chunk @ w2T ; wave w owns cols [w*64, w*64+64)
        for (int k0 = 0; k0 < 256; k0 += 32) {
            v16bf a0 = ld_frag(hc + 0  * HC_LD + k0, HC_LD);
            v16bf a1 = ld_frag(hc + 16 * HC_LD + k0, HC_LD);
            #pragma unroll
            for (int nt = 0; nt < 4; nt++) {
                v16bf bf = ld_frag(w2e + (size_t)(wave * 64 + nt * 16) * H_ + hc0 + k0, H_);
                yacc[0][nt] = wmma_bf16(a0, bf, yacc[0][nt]);
                yacc[1][nt] = wmma_bf16(a1, bf, yacc[1][nt]);
            }
        }
        __syncthreads();   // hc reused next chunk
    }

    // ---- epilogue: out[tok] (=/+=) w * (y + b2)
    #pragma unroll
    for (int mt = 0; mt < 2; mt++)
        #pragma unroll
        for (int nt = 0; nt < 4; nt++) {
            int n = wave * 64 + nt * 16 + nlo;
            float bv = b2[(size_t)e * O_ + n];
            #pragma unroll
            for (int i = 0; i < 8; i++) {
                int m = mt * 16 + half * 8 + i;
                int tok = stok[m];
                if (tok < 0) continue;
                float val = swt[m] * (yacc[mt][nt][i] + bv);
                size_t idx = (size_t)tok * O_ + n;
                if (pass) val += out[idx];
                out[idx] = val;
            }
        }
}

// -------------------------------------------------------- entropy finalize
__global__ void k_entf(const float* __restrict__ blockEnt, float* __restrict__ entOut) {
    if (threadIdx.x == 0) {
        float s = 0.f;
        for (int i = 0; i < NBLK; i++) s += blockEnt[i];
        entOut[0] = s / (float)B_;
    }
}

// ---------------------------------------------------------------------------
extern "C" void kernel_launch(void* const* d_in, const int* in_sizes, int n_in,
                              void* d_out, int out_size, void* d_ws, size_t ws_size,
                              hipStream_t stream) {
    const float* x      = (const float*)d_in[0];
    const float* gate_w = (const float*)d_in[1];
    const float* gate_b = (const float*)d_in[2];
    const float* w1     = (const float*)d_in[3];
    const float* b1     = (const float*)d_in[4];
    const float* w2     = (const float*)d_in[5];
    const float* b2     = (const float*)d_in[6];

    float* outY   = (float*)d_out;
    float* rawOut = outY + (size_t)B_ * O_;
    float* idxOut = rawOut + (size_t)B_ * E_;
    float* entOut = idxOut + (size_t)B_ * K_;

    char* ws = (char*)d_ws;
    float* blockEnt = (float*)(ws + 0);            // 64 f32
    int*   blockCnt = (int*)  (ws + 256);          // 64*16 i32
    int*   blockOff = (int*)  (ws + 4352);         // 64*16 i32
    int*   tileExp  = (int*)  (ws + 8448);         // 2*520 i32
    int*   tokE     = (int*)  (ws + 12608);        // B*2 i32
    float* tokW     = (float*)(ws + 143680);       // B*2 f32
    int*   slotTok  = (int*)  (ws + 274752);       // 2*SLOTCAP i32
    float* slotW    = (float*)(ws + 407872);       // 2*SLOTCAP f32
    ushort_t* xb    = (ushort_t*)(ws + 541184);    // B*D bf16   (16 MB)
    ushort_t* w1T   = (ushort_t*)(ws + 17318400);  // E*H*D bf16 (8 MB)
    ushort_t* w2T   = (ushort_t*)(ws + 25707008);  // E*O*H bf16 (8 MB)

    k_cvt_x<<<(B_ * D_) / (256 * 4), 256, 0, stream>>>(x, xb);
    k_transpose<<<E_ * (D_ / 32) * (H_ / 32), 256, 0, stream>>>(w1, w1T, D_, H_);
    k_transpose<<<E_ * (H_ / 32) * (O_ / 32), 256, 0, stream>>>(w2, w2T, H_, O_);

    k_route<<<NBLK, 256, 0, stream>>>(x, gate_w, gate_b, rawOut, idxOut,
                                      tokE, tokW, blockCnt, blockEnt);
    k_scan<<<1, 32, 0, stream>>>(blockCnt, blockOff, tileExp, slotTok);
    k_slot<<<NBLK, 256, 0, stream>>>(tokE, tokW, blockOff, slotTok, slotW);

    k_moe<<<TILES_K, 256, 0, stream>>>(0, slotTok, slotW, tileExp, xb, w1T, w2T, b1, b2, outY);
    k_moe<<<TILES_K, 256, 0, stream>>>(1, slotTok, slotW, tileExp, xb, w1T, w2T, b1, b2, outY);

    k_entf<<<1, 32, 0, stream>>>(blockEnt, entOut);
}
